// GRACE_18957985644564
// MI455X (gfx1250) — matrix operand order repaired and checked by reading, not compiled, not run on hardware
//
#include <hip/hip_runtime.h>
#include <hip/hip_bf16.h>

// ---------------------------------------------------------------------------
// Types for CDNA5 WMMA (wave32): v_wmma_f32_16x16x32_bf16
// ---------------------------------------------------------------------------
typedef __attribute__((ext_vector_type(16))) __bf16 vbf16x16;
typedef __attribute__((ext_vector_type(8)))  __bf16 vbf8;
typedef __attribute__((ext_vector_type(8)))  float  v8f;

__device__ __forceinline__ unsigned short f2bf_bits(float f) {
    // round-to-nearest-even fp32 -> bf16 bit pattern
    unsigned u = __float_as_uint(f);
    u += 0x7FFFu + ((u >> 16) & 1u);
    return (unsigned short)(u >> 16);
}

#define SHUF16(lo, hi) __builtin_shufflevector((lo), (hi), 0, 1, 2, 3, 4, 5, 6, 7, \
                                               8, 9, 10, 11, 12, 13, 14, 15)

// ---------------------------------------------------------------------------
// Degree kernels: deg starts at 1.0 (self-loop), atomic count, then rsqrt.
// ---------------------------------------------------------------------------
__global__ void deg_init_kernel(float* __restrict__ deg, int n) {
    int i = blockIdx.x * blockDim.x + threadIdx.x;
    if (i < n) deg[i] = 1.0f;
}

__global__ void deg_count_kernel(const int* __restrict__ dst,
                                 float* __restrict__ deg, int E) {
    int e = blockIdx.x * blockDim.x + threadIdx.x;
    if (e < E) {
        if (e + 8192 < E) __builtin_prefetch(dst + e + 8192, 0, 0);
        atomicAdd(&deg[dst[e]], 1.0f);
    }
}

__global__ void deg_rsqrt_kernel(float* __restrict__ deg, int n) {
    int i = blockIdx.x * blockDim.x + threadIdx.x;
    if (i < n) deg[i] = rsqrtf(deg[i]);   // deg >= 1 always (self-loop)
}

// ---------------------------------------------------------------------------
// fp32 -> bf16 streaming conversion (float4 in, 4x bf16 out per thread)
// ---------------------------------------------------------------------------
__global__ void cvt_bf16_kernel(const float* __restrict__ in,
                                unsigned short* __restrict__ out, long long n4) {
    long long t = (long long)blockIdx.x * blockDim.x + threadIdx.x;
    if (t >= n4) return;
    float4 v = reinterpret_cast<const float4*>(in)[t];
    ushort4 o;
    o.x = f2bf_bits(v.x); o.y = f2bf_bits(v.y);
    o.z = f2bf_bits(v.z); o.w = f2bf_bits(v.w);
    reinterpret_cast<ushort4*>(out)[t] = o;
}

// Fused PReLU + fp32 -> bf16 (layer-1 activation feeding GEMM2)
__global__ void prelu_cvt_kernel(const float* __restrict__ in,
                                 unsigned short* __restrict__ out, long long n4,
                                 const float* __restrict__ a_ptr) {
    long long t = (long long)blockIdx.x * blockDim.x + threadIdx.x;
    if (t >= n4) return;
    float a = *a_ptr;
    float4 v = reinterpret_cast<const float4*>(in)[t];
    v.x = (v.x >= 0.0f) ? v.x : a * v.x;
    v.y = (v.y >= 0.0f) ? v.y : a * v.y;
    v.z = (v.z >= 0.0f) ? v.z : a * v.z;
    v.w = (v.w >= 0.0f) ? v.w : a * v.w;
    ushort4 o;
    o.x = f2bf_bits(v.x); o.y = f2bf_bits(v.y);
    o.z = f2bf_bits(v.z); o.w = f2bf_bits(v.w);
    reinterpret_cast<ushort4*>(out)[t] = o;
}

// W[K=128 x F] fp32 -> Wt[F x 128] bf16 (transpose so B fragments are contiguous)
__global__ void wtrans_bf16_kernel(const float* __restrict__ W,
                                   unsigned short* __restrict__ Wtb, int F) {
    int t = blockIdx.x * blockDim.x + threadIdx.x;
    if (t >= F * 128) return;
    int f = t >> 7;        // row of Wt
    int k = t & 127;       // col of Wt
    Wtb[t] = f2bf_bits(W[(size_t)k * F + f]);
}

// ---------------------------------------------------------------------------
// WMMA GEMM: Y[n x F] = X[n x 128] @ W[128 x F]; X pre-converted to bf16
// row-major, W pre-converted to bf16 transposed (F x 128).
// blockDim = 128 (4 waves). Each wave computes a 64x16 C strip (4 M-tiles),
// reusing its B fragment across the 4 WMMAs of each 32-wide K slab.
// ---------------------------------------------------------------------------
__global__ __launch_bounds__(128)
void wmma_gemm_kernel(const __bf16* __restrict__ Xb,   // n x 128 bf16
                      const __bf16* __restrict__ Wtb,  // F x 128 bf16
                      float* __restrict__ Y,           // n x F fp32
                      int n, int F) {
    const int lane = threadIdx.x & 31;
    const int wave = threadIdx.x >> 5;
    const int row0 = blockIdx.x * 64;
    const int col0 = (blockIdx.y * 4 + wave) * 16;   // grid exactly covers F

    const int half = lane >> 4;     // 0: lanes 0-15, 1: lanes 16-31
    const int lid  = lane & 15;
    const int bcol = col0 + lid;
    const __bf16* bp = Wtb + (size_t)bcol * 128;

    // Clamped A-row pointers (out-of-range rows compute garbage for C rows
    // that are never stored) -> all loads unconditional, no exec juggling.
    const __bf16* ap[4];
    #pragma unroll
    for (int m = 0; m < 4; ++m) {
        int r = row0 + m * 16 + lid;
        if (r >= n) r = n - 1;
        ap[m] = Xb + (size_t)r * 128;
    }

    v8f acc[4] = {};
    #pragma unroll
    for (int kk = 0; kk < 128; kk += 32) {
        // B 32x16 fragment: lanes 0-15 hold K=kk+{0..7,16..23}, lanes 16-31 K=kk+{8..15,24..31}
        const int kb = kk + half * 8;
        vbf8 b_lo = *reinterpret_cast<const vbf8*>(bp + kb);
        vbf8 b_hi = *reinterpret_cast<const vbf8*>(bp + kb + 16);
        vbf16x16 bfr = SHUF16(b_lo, b_hi);

        // A 16x32 fragments: lanes 0-15 hold K=kk+{0..7,16..23},
        //                    lanes 16-31 hold K=kk+{8..15,24..31}
        const int ka = kk + half * 8;
        #pragma unroll
        for (int m = 0; m < 4; ++m) {
            vbf8 a_lo = *reinterpret_cast<const vbf8*>(ap[m] + ka);
            vbf8 a_hi = *reinterpret_cast<const vbf8*>(ap[m] + ka + 16);
            vbf16x16 afr = SHUF16(a_lo, a_hi);
            acc[m] = __builtin_amdgcn_wmma_f32_16x16x32_bf16(
                         false, afr, false, bfr, (short)0, acc[m], false, false);
        }
    }

    // C/D layout: VGPR r -> M = r (lanes 0-15) / 8+r (lanes 16-31), N = lid
    if (row0 + 64 <= n) {            // fast path: whole strip in range
        #pragma unroll
        for (int m = 0; m < 4; ++m) {
            const int crow = row0 + m * 16 + half * 8;
            #pragma unroll
            for (int r = 0; r < 8; ++r)
                Y[(size_t)(crow + r) * F + bcol] = acc[m][r];
        }
    } else {
        #pragma unroll
        for (int m = 0; m < 4; ++m) {
            const int crow = row0 + m * 16 + half * 8;
            #pragma unroll
            for (int r = 0; r < 8; ++r)
                if (crow + r < n) Y[(size_t)(crow + r) * F + bcol] = acc[m][r];
        }
    }
}

// ---------------------------------------------------------------------------
// agg[i][c] = bias[c] + h[i][c] * dinv[i]^2  (bias + self-loop, atomic-free)
// ---------------------------------------------------------------------------
template <int F>
__global__ void init_agg_kernel(const float* __restrict__ h,
                                const float* __restrict__ bias,
                                const float* __restrict__ dinv,
                                float* __restrict__ agg, int n) {
    constexpr int F4 = F / 4;
    long long t = (long long)blockIdx.x * blockDim.x + threadIdx.x;
    if (t >= (long long)n * F4) return;
    int r  = (int)(t / F4);
    int c4 = (int)(t % F4);
    float di = dinv[r];
    float s  = di * di;
    float4 hv = reinterpret_cast<const float4*>(h)[t];
    float4 bv = reinterpret_cast<const float4*>(bias)[c4];
    float4 o  = make_float4(bv.x + hv.x * s, bv.y + hv.y * s,
                            bv.z + hv.z * s, bv.w + hv.w * s);
    reinterpret_cast<float4*>(agg)[t] = o;
}

// ---------------------------------------------------------------------------
// Edge scatter: one thread = (edge, 4 channels). Coalesced float4 gather of
// h[src], scale by dinv[src]*dinv[dst], atomic-add into agg[dst] (L2-resident).
// ---------------------------------------------------------------------------
template <int F>
__global__ void edge_scatter_kernel(const int* __restrict__ src,
                                    const int* __restrict__ dst,
                                    const float* __restrict__ h,
                                    const float* __restrict__ dinv,
                                    float* __restrict__ agg, int E) {
    constexpr int F4 = F / 4;
    long long t = (long long)blockIdx.x * blockDim.x + threadIdx.x;
    if (t >= (long long)E * F4) return;
    int e = (int)(t / F4);
    int c = (int)(t % F4) << 2;

    if (c == 0 && e + 4096 < E) {   // stream edge lists ahead (global_prefetch_b8)
        __builtin_prefetch(src + e + 4096, 0, 0);
        __builtin_prefetch(dst + e + 4096, 0, 0);
    }

    int s = src[e], d = dst[e];
    float nrm = dinv[s] * dinv[d];
    const float4 hv = *reinterpret_cast<const float4*>(h + (size_t)s * F + c);
    float* out = agg + (size_t)d * F + c;
    atomicAdd(out + 0, hv.x * nrm);
    atomicAdd(out + 1, hv.y * nrm);
    atomicAdd(out + 2, hv.z * nrm);
    atomicAdd(out + 3, hv.w * nrm);
}

// Final in-place PReLU (fp32, float4-wide)
__global__ void prelu4_kernel(float* __restrict__ x, long long n4,
                              const float* __restrict__ a_ptr) {
    long long t = (long long)blockIdx.x * blockDim.x + threadIdx.x;
    if (t >= n4) return;
    float a = *a_ptr;
    float4 v = reinterpret_cast<float4*>(x)[t];
    v.x = (v.x >= 0.0f) ? v.x : a * v.x;
    v.y = (v.y >= 0.0f) ? v.y : a * v.y;
    v.z = (v.z >= 0.0f) ? v.z : a * v.z;
    v.w = (v.w >= 0.0f) ? v.w : a * v.w;
    reinterpret_cast<float4*>(x)[t] = v;
}

// ---------------------------------------------------------------------------
// Orchestration
// ---------------------------------------------------------------------------
extern "C" void kernel_launch(void* const* d_in, const int* in_sizes, int n_in,
                              void* d_out, int out_size, void* d_ws, size_t ws_size,
                              hipStream_t stream) {
    const int IN = 128, HID = 128, OUT = 64;

    const float* x   = (const float*)d_in[0];
    const int*   ei  = (const int*)  d_in[1];
    const float* W1  = (const float*)d_in[2];
    const float* b1  = (const float*)d_in[3];
    const float* W2  = (const float*)d_in[4];
    const float* b2  = (const float*)d_in[5];
    const float* a_p = (const float*)d_in[6];
    float* out = (float*)d_out;

    const int N = in_sizes[0] / IN;
    const int E = in_sizes[1] / 2;
    const int* src = ei;        // edge_index[0, :]
    const int* dst = ei + E;    // edge_index[1, :]

    // Scratch carve-out (256 B aligned regions)
    auto align256 = [](size_t v) { return (v + 255) & ~(size_t)255; };
    char* p = (char*)d_ws;
    float*          dinv = (float*)p;          p += align256((size_t)N * 4);
    unsigned short* Xb   = (unsigned short*)p; p += align256((size_t)N * HID * 2); // reused as h1b
    float*          h1   = (float*)p;          p += align256((size_t)N * HID * 4);
    float*          agg1 = (float*)p;          p += align256((size_t)N * HID * 4);
    float*          h2   = (float*)p;          p += align256((size_t)N * OUT * 4);
    unsigned short* W1tb = (unsigned short*)p; p += align256((size_t)HID * 128 * 2);
    unsigned short* W2tb = (unsigned short*)p; p += align256((size_t)OUT * 128 * 2);
    (void)ws_size; (void)n_in; (void)out_size;

    const int B = 256;
    auto nb = [](long long total, int blk) { return (unsigned)((total + blk - 1) / blk); };

    // 1) degrees -> dinv (in place)
    deg_init_kernel <<<nb(N, B), B, 0, stream>>>(dinv, N);
    deg_count_kernel<<<nb(E, B), B, 0, stream>>>(dst, dinv, E);
    deg_rsqrt_kernel<<<nb(N, B), B, 0, stream>>>(dinv, N);

    // 2) precision prep: X -> bf16, W1/W2 -> bf16 transposed
    cvt_bf16_kernel   <<<nb((long long)N * IN / 4, B), B, 0, stream>>>(x, Xb, (long long)N * IN / 4);
    wtrans_bf16_kernel<<<nb(HID * 128, B), B, 0, stream>>>(W1, W1tb, HID);
    wtrans_bf16_kernel<<<nb(OUT * 128, B), B, 0, stream>>>(W2, W2tb, OUT);

    // 3) Layer 1
    {
        dim3 grid((N + 63) / 64, HID / 64);
        wmma_gemm_kernel<<<grid, 128, 0, stream>>>((const __bf16*)Xb, (const __bf16*)W1tb, h1, N, HID);
    }
    init_agg_kernel<128>   <<<nb((long long)N * HID / 4, B), B, 0, stream>>>(h1, b1, dinv, agg1, N);
    edge_scatter_kernel<128><<<nb((long long)E * (HID / 4), B), B, 0, stream>>>(src, dst, h1, dinv, agg1, E);
    // fused PReLU + bf16 convert (reuse Xb region as layer-2 GEMM input)
    prelu_cvt_kernel<<<nb((long long)N * HID / 4, B), B, 0, stream>>>(agg1, Xb, (long long)N * HID / 4, a_p);

    // 4) Layer 2
    {
        dim3 grid((N + 63) / 64, 1);   // 4 waves cover OUT=64 columns
        wmma_gemm_kernel<<<grid, 128, 0, stream>>>((const __bf16*)Xb, (const __bf16*)W2tb, h2, N, OUT);
    }
    init_agg_kernel<64>    <<<nb((long long)N * OUT / 4, B), B, 0, stream>>>(h2, b2, dinv, out, N);
    edge_scatter_kernel<64><<<nb((long long)E * (OUT / 4), B), B, 0, stream>>>(src, dst, h2, dinv, out, E);
    prelu4_kernel<<<nb((long long)N * OUT / 4, B), B, 0, stream>>>(out, (long long)N * OUT / 4, a_p);
}
